// BaoNet_31559419691459
// MI455X (gfx1250) — matrix operand order repaired and checked by reading, not compiled, not run on hardware
//
#include <hip/hip_runtime.h>
#include <hip/hip_bf16.h>

// ---------------- problem constants (match reference) ----------------
#define N_NODES  100000
#define N_EDGES  3200000
#define N_GRAPHS 1024
#define IN_DIM   13
#define HID      64
#define OUT_DIM  72
#define N_LAYERS 4
#define M_TILES  (N_NODES / 16)         // 6250 exactly
#define SCAN_B   256
#define NB       ((N_NODES + SCAN_B - 1) / SCAN_B)   // 391 scan blocks

typedef __attribute__((ext_vector_type(16))) _Float16 v16h;
typedef __attribute__((ext_vector_type(8)))  float    v8f;

// ---------------------------------------------------------------------
__global__ void k_zero_f32(float* __restrict__ p, int n) {
    int i = blockIdx.x * blockDim.x + threadIdx.x;
    if (i < n) p[i] = 0.0f;
}
__global__ void k_zero_i32(int* __restrict__ p, int n) {
    int i = blockIdx.x * blockDim.x + threadIdx.x;
    if (i < n) p[i] = 0;
}
__global__ void k_copy_i32(const int* __restrict__ s, int* __restrict__ d, int n) {
    int i = blockIdx.x * blockDim.x + threadIdx.x;
    if (i < n) d[i] = s[i];
}

// ---------------------------------------------------------------------
// input projection: h = leaky_relu(Vnode @ W_in + b_in)   [N,13]@[13,64]
__global__ void k_input_proj(const float* __restrict__ Vnode,
                             const float* __restrict__ W_in,
                             const float* __restrict__ b_in,
                             float* __restrict__ h) {
    int idx = blockIdx.x * blockDim.x + threadIdx.x;
    if (idx >= N_NODES * HID) return;
    int n = idx >> 6;
    int j = idx & 63;
    const float* vrow = Vnode + (size_t)n * IN_DIM;
    float acc = b_in[j];
#pragma unroll
    for (int k = 0; k < IN_DIM; ++k)
        acc = fmaf(vrow[k], W_in[k * HID + j], acc);
    h[idx] = acc > 0.0f ? acc : 0.01f * acc;
}

// ---------------------------------------------------------------------
// pack weights into ISA B-matrix (32x16, 16-bit) fragment layout:
// lane L holds column n = L&15; halves i hold k = 32c + (L>=16?8:0) + (i&7) + (i>=8?16:0).
__global__ void k_pack_weights(const float* __restrict__ Wself,
                               const float* __restrict__ Wnbr,
                               _Float16* __restrict__ wpk) {
    int tid = blockIdx.x * blockDim.x + threadIdx.x;
    if (tid >= N_LAYERS * 2 * 4 * 2 * 32 * 16) return;
    int i    =  tid        & 15;
    int lane = (tid >> 4)  & 31;
    int c    = (tid >> 9)  & 1;
    int nt   = (tid >> 10) & 3;
    int mat  = (tid >> 12) & 1;
    int l    = (tid >> 13) & 3;
    int k = 32 * c + ((lane >= 16) ? 8 : 0) + (i & 7) + ((i >= 8) ? 16 : 0);
    int n = nt * 16 + (lane & 15);
    const float* W = (mat == 0 ? Wself : Wnbr) + (size_t)l * HID * HID;
    wpk[tid] = (_Float16)W[k * HID + n];
}

// ---------------------------------------------------------------------
// CSR build step 1: in-degree histogram of dst
__global__ void k_hist(const long long* __restrict__ dst, int* __restrict__ count) {
    int e = blockIdx.x * blockDim.x + threadIdx.x;
    if (e >= N_EDGES) return;
    __builtin_prefetch(dst + e + 8192, 0, 1);   // global_prefetch_b8
    atomicAdd(count + (int)dst[e], 1);
}

// CSR build step 2a: per-block exclusive scan of counts (Hillis-Steele)
__global__ void k_scan1(const int* __restrict__ count,
                        int* __restrict__ excl, int* __restrict__ bsum) {
    __shared__ int sh[SCAN_B];
    int t = threadIdx.x;
    int i = blockIdx.x * SCAN_B + t;
    int v = (i < N_NODES) ? count[i] : 0;
    sh[t] = v;
    __syncthreads();
#pragma unroll
    for (int off = 1; off < SCAN_B; off <<= 1) {
        int x = (t >= off) ? sh[t - off] : 0;
        __syncthreads();
        if (t >= off) sh[t] += x;
        __syncthreads();
    }
    if (i < N_NODES) excl[i] = sh[t] - v;            // exclusive
    if (t == SCAN_B - 1) bsum[blockIdx.x] = sh[t];   // block total
}

// CSR build step 2b: scan the NB (=391) block totals in one 512-thread block
__global__ void k_scan2(int* __restrict__ bsum) {
    __shared__ int sh[512];
    int t = threadIdx.x;
    int v = (t < NB) ? bsum[t] : 0;
    sh[t] = v;
    __syncthreads();
#pragma unroll
    for (int off = 1; off < 512; off <<= 1) {
        int x = (t >= off) ? sh[t - off] : 0;
        __syncthreads();
        if (t >= off) sh[t] += x;
        __syncthreads();
    }
    if (t < NB) bsum[t] = sh[t] - v;                 // exclusive block offsets
}

// CSR build step 2c: rowptr = excl + block offset
__global__ void k_scan3(const int* __restrict__ excl, const int* __restrict__ bsum,
                        int* __restrict__ rowptr) {
    int i = blockIdx.x * blockDim.x + threadIdx.x;
    if (i < N_NODES) rowptr[i] = excl[i] + bsum[i / SCAN_B];
}

// CSR build step 3: bucket-fill edge sources grouped by dst (int32 indices)
__global__ void k_fill(const long long* __restrict__ src,
                       const long long* __restrict__ dst,
                       int* __restrict__ cursor, int* __restrict__ eidx) {
    int e = blockIdx.x * blockDim.x + threadIdx.x;
    if (e >= N_EDGES) return;
    __builtin_prefetch(src + e + 8192, 0, 1);
    __builtin_prefetch(dst + e + 8192, 0, 1);
    int d = (int)dst[e];
    int pos = atomicAdd(cursor + d, 1);
    eidx[pos] = (int)src[e];
}

// ---------------------------------------------------------------------
// atomic-free message aggregation: one wave32 per node, lane = channel pair.
// Neighbor indices loaded coalesced (one per lane) and broadcast via shfl.
__global__ void k_gather_csr(const float* __restrict__ h,
                             const int* __restrict__ rowptr,
                             const int* __restrict__ count,
                             const int* __restrict__ eidx,
                             float* __restrict__ msg) {
    int wave = blockIdx.x * (blockDim.x >> 5) + (threadIdx.x >> 5);
    if (wave >= N_NODES) return;
    int lane = threadIdx.x & 31;
    int beg = rowptr[wave];
    int end = beg + count[wave];
    float a0 = 0.0f, a1 = 0.0f;
    for (int base = beg; base < end; base += 32) {
        int idx = base + lane;
        int s = (idx < end) ? eidx[idx] : 0;
        int cnt = end - base; if (cnt > 32) cnt = 32;
        for (int j = 0; j < cnt; ++j) {
            int sj = __shfl(s, j, 32);
            const float* hr = h + (size_t)sj * HID;
            a0 += hr[lane];
            a1 += hr[lane + 32];
        }
    }
    msg[(size_t)wave * HID + lane]      = a0;
    msg[(size_t)wave * HID + lane + 32] = a1;
}

// ---------------------------------------------------------------------
__device__ __forceinline__ v16h cvt16(const float* __restrict__ p0,
                                      const float* __restrict__ p1) {
    float4 a = ((const float4*)p0)[0];
    float4 b = ((const float4*)p0)[1];
    float4 c = ((const float4*)p1)[0];
    float4 d = ((const float4*)p1)[1];
    v16h v;
    v[0]  = (_Float16)a.x; v[1]  = (_Float16)a.y; v[2]  = (_Float16)a.z; v[3]  = (_Float16)a.w;
    v[4]  = (_Float16)b.x; v[5]  = (_Float16)b.y; v[6]  = (_Float16)b.z; v[7]  = (_Float16)b.w;
    v[8]  = (_Float16)c.x; v[9]  = (_Float16)c.y; v[10] = (_Float16)c.z; v[11] = (_Float16)c.w;
    v[12] = (_Float16)d.x; v[13] = (_Float16)d.y; v[14] = (_Float16)d.z; v[15] = (_Float16)d.w;
    return v;
}

// hnext = leaky_relu(h @ Wself + msg @ Wnbr + b); one wave per 16-node tile.
// A layout (16-bit 16x32): lane row = lane&15, koff = (lane>=16)?8:0,
// halves 0..7 -> K = 32c+koff+i, halves 8..15 -> K = 32c+16+koff+(i-8).
__global__ void k_layer_gemm(const float* __restrict__ h,
                             const float* __restrict__ msg,
                             const _Float16* __restrict__ wpk_l, // 8192 halves
                             const float* __restrict__ bias,     // 64
                             float* __restrict__ hout) {
    int wave = blockIdx.x * (blockDim.x >> 5) + (threadIdx.x >> 5);
    if (wave >= M_TILES) return;               // wave-uniform: EXEC stays all-1s
    int lane = threadIdx.x & 31;
    int half = lane >> 4;
    int lm   = lane & 15;
    int koff = half * 8;
    size_t mbase = (size_t)wave * 16;
    const float* hrow = h   + (mbase + lm) * HID;
    const float* mrow = msg + (mbase + lm) * HID;

    v16h ah0 = cvt16(hrow + koff,      hrow + 16 + koff);
    v16h ah1 = cvt16(hrow + 32 + koff, hrow + 48 + koff);
    v16h am0 = cvt16(mrow + koff,      mrow + 16 + koff);
    v16h am1 = cvt16(mrow + 32 + koff, mrow + 48 + koff);

    const _Float16* wself = wpk_l;
    const _Float16* wnbr  = wpk_l + 4096;

#pragma unroll
    for (int nt = 0; nt < 4; ++nt) {
        v16h b0 = *(const v16h*)(wself + (size_t)((nt * 2 + 0) * 32 + lane) * 16);
        v16h b1 = *(const v16h*)(wself + (size_t)((nt * 2 + 1) * 32 + lane) * 16);
        v16h c0 = *(const v16h*)(wnbr  + (size_t)((nt * 2 + 0) * 32 + lane) * 16);
        v16h c1 = *(const v16h*)(wnbr  + (size_t)((nt * 2 + 1) * 32 + lane) * 16);
        v8f acc = {};
        acc = __builtin_amdgcn_wmma_f32_16x16x32_f16(false, ah0, false, b0, (short)0, acc, false, false);
        acc = __builtin_amdgcn_wmma_f32_16x16x32_f16(false, ah1, false, b1, (short)0, acc, false, false);
        acc = __builtin_amdgcn_wmma_f32_16x16x32_f16(false, am0, false, c0, (short)0, acc, false, false);
        acc = __builtin_amdgcn_wmma_f32_16x16x32_f16(false, am1, false, c1, (short)0, acc, false, false);
        float bv = bias[nt * 16 + lm];
#pragma unroll
        for (int r = 0; r < 8; ++r) {
            float v = acc[r] + bv;
            v = v > 0.0f ? v : 0.01f * v;
            hout[(mbase + r + half * 8) * HID + nt * 16 + lm] = v;
        }
    }
}

// ---------------------------------------------------------------------
// graph pooling (y sorted but tiny: atomics fine): one wave per node
__global__ void k_pool(const float* __restrict__ h,
                       const long long* __restrict__ y,
                       float* __restrict__ gsum,
                       float* __restrict__ gcnt) {
    int tid  = blockIdx.x * blockDim.x + threadIdx.x;
    int n    = tid >> 5;
    int lane = tid & 31;
    if (n >= N_NODES) return;
    int g = (int)y[n];
    const float* hr = h + (size_t)n * HID;
    float* gs = gsum + (size_t)g * HID;
    atomicAdd(gs + lane,      hr[lane]);
    atomicAdd(gs + lane + 32, hr[lane + 32]);
    if (lane == 0) atomicAdd(gcnt + g, 1.0f);
}

// ---------------------------------------------------------------------
__global__ void k_head(const float* __restrict__ gsum,
                       const float* __restrict__ gcnt,
                       const float* __restrict__ Wout, const float* __restrict__ bout,
                       const float* __restrict__ W1,   const float* __restrict__ b1,
                       const float* __restrict__ W2,   const float* __restrict__ b2,
                       float* __restrict__ out) {
    __shared__ float shA[72];
    __shared__ float shB[72];
    int g   = blockIdx.x;
    int tid = threadIdx.x;
    if (tid < HID) {
        float c = gcnt[g];
        c = c > 1.0f ? c : 1.0f;
        shA[tid] = gsum[(size_t)g * HID + tid] / c;
    }
    __syncthreads();
    if (tid < OUT_DIM) {
        float acc = bout[tid];
#pragma unroll
        for (int k = 0; k < HID; ++k)
            acc = fmaf(shA[k], Wout[k * OUT_DIM + tid], acc);
        shB[tid] = acc;
    }
    __syncthreads();
    if (tid < 36) {
        float acc = b1[tid];
#pragma unroll
        for (int k = 0; k < OUT_DIM; ++k)
            acc = fmaf(shB[k], W1[k * 36 + tid], acc);
        shA[tid] = acc > 0.0f ? acc : 0.01f * acc;
    }
    __syncthreads();
    if (tid == 0) {
        float acc = b2[0];
#pragma unroll
        for (int k = 0; k < 36; ++k)
            acc = fmaf(shA[k], W2[k], acc);
        out[g] = acc;
    }
}

// ---------------------------------------------------------------------
extern "C" void kernel_launch(void* const* d_in, const int* in_sizes, int n_in,
                              void* d_out, int out_size, void* d_ws, size_t ws_size,
                              hipStream_t stream) {
    const float*     Vnode = (const float*)d_in[0];
    const long long* Vedge = (const long long*)d_in[1];
    const long long* y     = (const long long*)d_in[2];
    const float*     W_in  = (const float*)d_in[3];
    const float*     b_in  = (const float*)d_in[4];
    const float*     Wself = (const float*)d_in[5];
    const float*     Wnbr  = (const float*)d_in[6];
    const float*     bl    = (const float*)d_in[7];
    const float*     Wout  = (const float*)d_in[8];
    const float*     bout  = (const float*)d_in[9];
    const float*     W1    = (const float*)d_in[10];
    const float*     b1    = (const float*)d_in[11];
    const float*     W2    = (const float*)d_in[12];
    const float*     b2    = (const float*)d_in[13];
    float* out = (float*)d_out;

    const long long* esrc = Vedge;
    const long long* edst = Vedge + N_EDGES;

    char* ws = (char*)d_ws;
    size_t off = 0;
    auto carve = [&](size_t bytes) -> char* {
        char* p = ws + off;
        off = (off + bytes + 255) & ~(size_t)255;
        return p;
    };
    float*    hA     = (float*)    carve((size_t)N_NODES * HID * 4);
    float*    hB     = (float*)    carve((size_t)N_NODES * HID * 4);
    float*    msg    = (float*)    carve((size_t)N_NODES * HID * 4);
    _Float16* wpk    = (_Float16*) carve((size_t)N_LAYERS * 8192 * 2);
    float*    gsum   = (float*)    carve((size_t)N_GRAPHS * HID * 4);
    float*    gcnt   = (float*)    carve((size_t)N_GRAPHS * 4);
    int*      count  = (int*)      carve((size_t)N_NODES * 4);
    int*      excl   = (int*)      carve((size_t)N_NODES * 4);
    int*      bsum   = (int*)      carve((size_t)512 * 4);
    int*      rowptr = (int*)      carve((size_t)N_NODES * 4);
    int*      cursor = (int*)      carve((size_t)N_NODES * 4);
    int*      eidx   = (int*)      carve((size_t)N_EDGES * 4);
    (void)ws_size; (void)n_in; (void)in_sizes; (void)out_size;

    // ---- init accumulators (ws is poisoned each run)
    k_zero_f32<<<(N_GRAPHS * HID + 255) / 256, 256, 0, stream>>>(gsum, N_GRAPHS * HID);
    k_zero_f32<<<(N_GRAPHS + 255) / 256, 256, 0, stream>>>(gcnt, N_GRAPHS);
    k_zero_i32<<<(N_NODES + 255) / 256, 256, 0, stream>>>(count, N_NODES);

    // ---- CSR build (edge structure is reused by all 4 layers)
    k_hist<<<(N_EDGES + 255) / 256, 256, 0, stream>>>(edst, count);
    k_scan1<<<NB, SCAN_B, 0, stream>>>(count, excl, bsum);
    k_scan2<<<1, 512, 0, stream>>>(bsum);
    k_scan3<<<NB, SCAN_B, 0, stream>>>(excl, bsum, rowptr);
    k_copy_i32<<<NB, SCAN_B, 0, stream>>>(rowptr, cursor, N_NODES);
    k_fill<<<(N_EDGES + 255) / 256, 256, 0, stream>>>(esrc, edst, cursor, eidx);

    // ---- weight fragment packing + input projection
    k_pack_weights<<<(N_LAYERS * 2 * 4 * 2 * 32 * 16 + 255) / 256, 256, 0, stream>>>(Wself, Wnbr, wpk);
    k_input_proj<<<(N_NODES * HID + 255) / 256, 256, 0, stream>>>(Vnode, W_in, b_in, hA);

    // ---- message-passing layers
    float* hcur = hA;
    float* hnxt = hB;
    for (int l = 0; l < N_LAYERS; ++l) {
        k_gather_csr<<<(N_NODES + 7) / 8, 256, 0, stream>>>(hcur, rowptr, count, eidx, msg);
        k_layer_gemm<<<(M_TILES + 7) / 8, 256, 0, stream>>>(
            hcur, msg, wpk + (size_t)l * 8192, bl + (size_t)l * HID, hnxt);
        float* t = hcur; hcur = hnxt; hnxt = t;
    }

    // ---- pooling + MLP head
    k_pool<<<(N_NODES * 32 + 255) / 256, 256, 0, stream>>>(hcur, y, gsum, gcnt);
    k_head<<<N_GRAPHS, 128, 0, stream>>>(gsum, gcnt, Wout, bout, W1, b1, W2, b2, out);
}
